// ReprogrammingLayer_MQA_35622458753685
// MI455X (gfx1250) — matrix operand order repaired and checked by reading, not compiled, not run on hardware
//
#include <hip/hip_runtime.h>

// ---------------------------------------------------------------------------
// CDNA5 (gfx1250) mixed-precision implementation of the MQA reprogramming
// layer.  fp32 inputs are cast once to bf16; every matmul runs on
// v_wmma_f32_16x16x32_bf16 (fp32 accumulate); softmax is fp32.
// ---------------------------------------------------------------------------

typedef __attribute__((ext_vector_type(16))) __bf16 v16bf;
typedef __attribute__((ext_vector_type(8)))  float  v8f;
typedef __attribute__((ext_vector_type(4)))  unsigned int v4u;

union Frag {                 // 16 bf16 = 32 bytes = two 16B chunks
    v16bf v;
    v4u   q[2];
};

__device__ inline v8f zero8() {
    v8f z;
    #pragma unroll
    for (int i = 0; i < 8; ++i) z[i] = 0.f;
    return z;
}

__device__ inline v8f wmma_bf(v16bf a, v16bf b, v8f c) {
    // (neg_a, A, neg_b, B, c_mod, C, reuse_a, reuse_b)
    return __builtin_amdgcn_wmma_f32_16x16x32_bf16(false, a, false, b,
                                                   (short)0, c, false, false);
}

// ---------------------------------------------------------------------------
// fp32 -> bf16 casts / padded casts / transposed casts (one-time, BW-bound)
// ---------------------------------------------------------------------------
__global__ __launch_bounds__(256)
void cvt_f32_to_bf16(const float* __restrict__ in, __bf16* __restrict__ out) {
    size_t i = (size_t)blockIdx.x * 256 + threadIdx.x;
    out[i] = (__bf16)in[i];
}

// out has prow*cols elements (grid sized exactly); rows >= `rows` are zero
__global__ __launch_bounds__(256)
void cvt_pad_rows(const float* __restrict__ in, __bf16* __restrict__ out,
                  int rows, int cols) {
    size_t i = (size_t)blockIdx.x * 256 + threadIdx.x;
    int r = (int)(i / (size_t)cols);
    out[i] = (r < rows) ? (__bf16)in[i] : (__bf16)0.f;
}

// in is [R][C] fp32, out is [C][R] bf16
__global__ __launch_bounds__(256)
void transpose_cvt(const float* __restrict__ in, __bf16* __restrict__ out,
                   int R, int C) {
    size_t i = (size_t)blockIdx.x * 256 + threadIdx.x;
    int r = (int)(i / (size_t)C);
    int c = (int)(i % (size_t)C);
    out[(size_t)c * R + r] = (__bf16)in[i];
}

// ---------------------------------------------------------------------------
// Generic bf16 GEMM:  C[M,N] = A[M,K] * B[K,N] + bias[N]
//   A row-major bf16, B given pre-transposed (Bt[N][K] bf16).
//   Block tile 128x128, K-step 32, 8 waves -> each wave 64x32 (4x2 WMMA tiles)
//   OUT_MODE: 0 = fp32 out, 1 = bf16 out, 2 = bf16 transposed out (Out[N][M])
// ---------------------------------------------------------------------------
template <int OUT_MODE>
__global__ __launch_bounds__(256)
void gemm_bf16_128x128(const __bf16* __restrict__ A,
                       const __bf16* __restrict__ Bt,
                       const float* __restrict__ bias,
                       void* __restrict__ Out,
                       int M, int N, int K) {
    __shared__ __bf16 lA[128 * 32];
    __shared__ __bf16 lB[128 * 32];

    const int tid  = threadIdx.x;
    const int w    = tid >> 5;
    const int lane = tid & 31;
    const int lm   = lane & 15;   // row/col within a 16-wide tile
    const int g    = lane >> 4;   // half-wave group
    const int wm   = w >> 2;      // 0..1  -> 64-row strip
    const int wn   = w & 3;       // 0..3  -> 32-col strip
    const int m0   = blockIdx.y * 128;
    const int n0   = blockIdx.x * 128;

    v8f acc[4][2];
    #pragma unroll
    for (int mt = 0; mt < 4; ++mt)
        #pragma unroll
        for (int nt = 0; nt < 2; ++nt) acc[mt][nt] = zero8();

    for (int kb = 0; kb < K; kb += 32) {
        __syncthreads();
        // stage A[128x32] and Bt[128x32] (each 8KB) with 16B loads
        #pragma unroll
        for (int i = 0; i < 2; ++i) {
            int c   = tid + (i << 8);     // 0..511
            int row = c >> 2;
            int c8  = (c & 3) << 3;
            *(v4u*)&lA[row * 32 + c8] =
                *(const v4u*)&A[(size_t)(m0 + row) * K + kb + c8];
            *(v4u*)&lB[row * 32 + c8] =
                *(const v4u*)&Bt[(size_t)(n0 + row) * K + kb + c8];
        }
        __syncthreads();

        Frag a[4], b[2];
        #pragma unroll
        for (int mt = 0; mt < 4; ++mt) {
            const __bf16* p = &lA[(wm * 64 + mt * 16 + lm) * 32];
            a[mt].q[0] = *(const v4u*)&p[g * 8];        // K = g*8 .. g*8+7
            a[mt].q[1] = *(const v4u*)&p[16 + g * 8];   // K = 16+g*8 ..
        }
        #pragma unroll
        for (int nt = 0; nt < 2; ++nt) {
            const __bf16* p = &lB[(wn * 32 + nt * 16 + lm) * 32];
            b[nt].q[0] = *(const v4u*)&p[g * 8];        // K = g*8 .. g*8+7
            b[nt].q[1] = *(const v4u*)&p[16 + g * 8];   // K = 16+g*8 ..
        }
        #pragma unroll
        for (int mt = 0; mt < 4; ++mt)
            #pragma unroll
            for (int nt = 0; nt < 2; ++nt)
                acc[mt][nt] = wmma_bf(a[mt].v, b[nt].v, acc[mt][nt]);
    }

    // epilogue: C layout -> lane holds col n = lm, rows r + 8*g
    #pragma unroll
    for (int mt = 0; mt < 4; ++mt)
        #pragma unroll
        for (int nt = 0; nt < 2; ++nt)
            #pragma unroll
            for (int r = 0; r < 8; ++r) {
                int m = m0 + wm * 64 + mt * 16 + r + 8 * g;
                int n = n0 + wn * 32 + nt * 16 + lm;
                float val = acc[mt][nt][r] + bias[n];
                if (OUT_MODE == 0)
                    ((float*)Out)[(size_t)m * N + n] = val;
                else if (OUT_MODE == 1)
                    ((__bf16*)Out)[(size_t)m * N + n] = (__bf16)val;
                else
                    ((__bf16*)Out)[(size_t)n * M + m] = (__bf16)val;
            }
}

// ---------------------------------------------------------------------------
// Flash-style MQA attention.
//   q   : [B*L, H*E] bf16      (row = b*1024 + l, col = h*128 + e)
//   k   : [Spad=1024, E=128] bf16 (rows >= Sreal masked with -inf scores)
//   vT  : [E=128, Spad=1024] bf16
//   attn: [B*L, H*E] bf16 output
// Block = 256 threads (8 waves) handles one (b, h, 128-query-row tile).
// Each wave owns 16 query rows; loops over 8 S-tiles of 128.
// ---------------------------------------------------------------------------
__global__ __launch_bounds__(256)
void attn_flash(const __bf16* __restrict__ q, const __bf16* __restrict__ kk,
                const __bf16* __restrict__ vT, __bf16* __restrict__ attn,
                int Sreal) {
    extern __shared__ __bf16 smem[];
    __bf16* lK = smem;              // [128][128]  (s_local, e)
    __bf16* lV = smem + 16384;      // [128][128]  (e, s_local)
    __bf16* lP = smem + 32768;      // [128][128]  (m, s_local) wave-private rows

    const int tid  = threadIdx.x;
    const int w    = tid >> 5;
    const int lane = tid & 31;
    const int lm   = lane & 15;
    const int g    = lane >> 4;
    const int lt   = blockIdx.x;    // query tile
    const int h    = blockIdx.y;
    const int b    = blockIdx.z;
    const size_t row0 = (size_t)b * 1024 + (size_t)lt * 128;
    const float NEG_INF = -__builtin_inff();

    // Q fragments for this wave's 16 rows (kept resident in VGPRs)
    Frag aq[4];
    {
        const __bf16* qp = q + (row0 + 16 * w + lm) * 1024 + h * 128;
        #pragma unroll
        for (int ks = 0; ks < 4; ++ks) {
            aq[ks].q[0] = *(const v4u*)&qp[ks * 32 + g * 8];
            aq[ks].q[1] = *(const v4u*)&qp[ks * 32 + 16 + g * 8];
        }
    }

    float rowmax[8], rowsum[8];
    v8f o[8];
    #pragma unroll
    for (int r = 0; r < 8; ++r) { rowmax[r] = NEG_INF; rowsum[r] = 0.f; }
    #pragma unroll
    for (int et = 0; et < 8; ++et) o[et] = zero8();

    for (int st = 0; st < 8; ++st) {
        __syncthreads();
        // stage K tile [128s x 128e] and V^T tile [128e x 128s] (32KB each)
        #pragma unroll
        for (int i = 0; i < 8; ++i) {
            int c   = tid + (i << 8);          // 0..2047 chunks of 8 bf16
            int row = c >> 4;
            int c8  = (c & 15) << 3;
            *(v4u*)&lK[row * 128 + c8] =
                *(const v4u*)&kk[(size_t)(st * 128 + row) * 128 + c8];
            *(v4u*)&lV[row * 128 + c8] =
                *(const v4u*)&vT[(size_t)row * 1024 + st * 128 + c8];
        }
        __syncthreads();

        // scores: S = Q * K^T  (B-frag column n -> row of K, contiguous e)
        v8f sc[8];
        #pragma unroll
        for (int nt = 0; nt < 8; ++nt) {
            sc[nt] = zero8();
            const __bf16* kp = &lK[(nt * 16 + lm) * 128];
            #pragma unroll
            for (int ks = 0; ks < 4; ++ks) {
                Frag bfr;
                bfr.q[0] = *(const v4u*)&kp[ks * 32 + g * 8];
                bfr.q[1] = *(const v4u*)&kp[ks * 32 + 16 + g * 8];
                sc[nt] = wmma_bf(aq[ks].v, bfr.v, sc[nt]);
            }
        }

        // scale + mask s >= Sreal
        const float scale = 0.088388347648318447f;   // 1/sqrt(128)
        #pragma unroll
        for (int nt = 0; nt < 8; ++nt) {
            bool valid = (st * 128 + nt * 16 + lm) < Sreal;
            #pragma unroll
            for (int r = 0; r < 8; ++r)
                sc[nt][r] = valid ? sc[nt][r] * scale : NEG_INF;
        }

        // online softmax: each row's 16 columns live in a 16-lane group
        #pragma unroll
        for (int r = 0; r < 8; ++r) {
            float mx = sc[0][r];
            #pragma unroll
            for (int nt = 1; nt < 8; ++nt) mx = fmaxf(mx, sc[nt][r]);
            #pragma unroll
            for (int off = 1; off < 16; off <<= 1)
                mx = fmaxf(mx, __shfl_xor(mx, off, 32));
            float nm   = fmaxf(rowmax[r], mx);
            float corr = __expf(rowmax[r] - nm);
            rowmax[r]  = nm;
            float ps = 0.f;
            #pragma unroll
            for (int nt = 0; nt < 8; ++nt) {
                float p = __expf(sc[nt][r] - nm);
                sc[nt][r] = p;
                ps += p;
            }
            #pragma unroll
            for (int off = 1; off < 16; off <<= 1)
                ps += __shfl_xor(ps, off, 32);
            rowsum[r] = rowsum[r] * corr + ps;
            #pragma unroll
            for (int et = 0; et < 8; ++et) o[et][r] *= corr;
        }

        // P (C-layout) -> LDS (row-major) so it can be re-read in A-layout
        #pragma unroll
        for (int nt = 0; nt < 8; ++nt)
            #pragma unroll
            for (int r = 0; r < 8; ++r)
                lP[(16 * w + r + 8 * g) * 128 + nt * 16 + lm] =
                    (__bf16)sc[nt][r];
        // wave-private region: only need this wave's DS stores visible
        asm volatile("s_wait_dscnt 0x0" ::: "memory");

        // O += P * V   (B-frag column n -> row e of V^T, contiguous s)
        #pragma unroll
        for (int ks = 0; ks < 4; ++ks) {
            Frag ap;
            const __bf16* pp = &lP[(16 * w + lm) * 128 + ks * 32];
            ap.q[0] = *(const v4u*)&pp[g * 8];
            ap.q[1] = *(const v4u*)&pp[16 + g * 8];
            #pragma unroll
            for (int et = 0; et < 8; ++et) {
                Frag bv;
                const __bf16* vp = &lV[(et * 16 + lm) * 128 + ks * 32];
                bv.q[0] = *(const v4u*)&vp[g * 8];
                bv.q[1] = *(const v4u*)&vp[16 + g * 8];
                o[et] = wmma_bf(ap.v, bv.v, o[et]);
            }
        }
    }

    // normalize and store bf16
    #pragma unroll
    for (int r = 0; r < 8; ++r) {
        float inv = 1.0f / rowsum[r];
        size_t row = row0 + 16 * w + r + 8 * g;
        #pragma unroll
        for (int et = 0; et < 8; ++et) {
            int col = h * 128 + et * 16 + lm;
            attn[row * 1024 + col] = (__bf16)(o[et][r] * inv);
        }
    }
}

// ---------------------------------------------------------------------------
// launcher
// ---------------------------------------------------------------------------
extern "C" void kernel_launch(void* const* d_in, const int* in_sizes, int n_in,
                              void* d_out, int out_size, void* d_ws,
                              size_t ws_size, hipStream_t stream) {
    const float* tgt = (const float*)d_in[0];   // [8,1024,1024]
    const float* src = (const float*)d_in[1];   // [1000,4096]
    const float* val = (const float*)d_in[2];   // [1000,4096]
    const float* Wq  = (const float*)d_in[3];   // [1024,1024]
    const float* bq  = (const float*)d_in[4];
    const float* Wk  = (const float*)d_in[5];   // [4096,128]
    const float* bk  = (const float*)d_in[6];
    const float* Wv  = (const float*)d_in[7];   // [4096,128]
    const float* bv  = (const float*)d_in[8];
    const float* Wo  = (const float*)d_in[9];   // [1024,4096]
    const float* bo  = (const float*)d_in[10];
    float* out = (float*)d_out;                 // [8192,4096] fp32

    char*  ws  = (char*)d_ws;
    size_t off = 0;
    auto alloc = [&](size_t bytes) -> char* {
        char* p = ws + off;
        off += (bytes + 255) & ~(size_t)255;
        return p;
    };
    __bf16* tgtb = (__bf16*)alloc((size_t)8192 * 1024 * 2);  // target bf16
    __bf16* WqT  = (__bf16*)alloc((size_t)1024 * 1024 * 2);  // [N=1024][K=1024]
    __bf16* srcp = (__bf16*)alloc((size_t)1024 * 4096 * 2);  // padded source
    __bf16* valp = (__bf16*)alloc((size_t)1024 * 4096 * 2);  // padded value
    __bf16* WkT  = (__bf16*)alloc((size_t)128 * 4096 * 2);   // [128][4096]
    __bf16* WvT  = (__bf16*)alloc((size_t)128 * 4096 * 2);
    __bf16* WoT  = (__bf16*)alloc((size_t)4096 * 1024 * 2);  // [4096][1024]
    __bf16* qb   = (__bf16*)alloc((size_t)8192 * 1024 * 2);  // q  [8192][1024]
    __bf16* kb_  = (__bf16*)alloc((size_t)1024 * 128 * 2);   // k  [1024][128]
    __bf16* vTb  = (__bf16*)alloc((size_t)128 * 1024 * 2);   // v^T[128][1024]
    __bf16* attb = (__bf16*)alloc((size_t)8192 * 1024 * 2);  // attn out bf16

    // 1) casts / transposed casts (bandwidth-side, tiny vs GEMMs)
    cvt_f32_to_bf16<<<(8192 * 1024) / 256, 256, 0, stream>>>(tgt, tgtb);
    transpose_cvt<<<(1024 * 1024) / 256, 256, 0, stream>>>(Wq, WqT, 1024, 1024);
    cvt_pad_rows<<<(1024 * 4096) / 256, 256, 0, stream>>>(src, srcp, 1000, 4096);
    cvt_pad_rows<<<(1024 * 4096) / 256, 256, 0, stream>>>(val, valp, 1000, 4096);
    transpose_cvt<<<(4096 * 128) / 256, 256, 0, stream>>>(Wk, WkT, 4096, 128);
    transpose_cvt<<<(4096 * 128) / 256, 256, 0, stream>>>(Wv, WvT, 4096, 128);
    transpose_cvt<<<(1024 * 4096) / 256, 256, 0, stream>>>(Wo, WoT, 1024, 4096);

    // 2) projections (bf16 WMMA, fp32 accumulate)
    gemm_bf16_128x128<1><<<dim3(1024 / 128, 8192 / 128), 256, 0, stream>>>(
        tgtb, WqT, bq, qb, 8192, 1024, 1024);
    gemm_bf16_128x128<1><<<dim3(1, 1024 / 128), 256, 0, stream>>>(
        srcp, WkT, bk, kb_, 1024, 128, 4096);
    gemm_bf16_128x128<2><<<dim3(1, 1024 / 128), 256, 0, stream>>>(
        valp, WvT, bv, vTb, 1024, 128, 4096);    // stored transposed

    // 3) flash attention: grid (ltile=8, H=8, B=8), 96KB dynamic LDS
    attn_flash<<<dim3(8, 8, 8), 256, 96 * 1024, stream>>>(qb, kb_, vTb, attb,
                                                          1000);

    // 4) output projection + bias -> fp32 d_out
    gemm_bf16_128x128<0><<<dim3(4096 / 128, 8192 / 128), 256, 0, stream>>>(
        attb, WoT, bo, out, 8192, 4096, 1024);
}
